// LatentCodeFormer_87952340288150
// MI455X (gfx1250) — matrix operand order repaired and checked by reading, not compiled, not run on hardware
//
#include <hip/hip_runtime.h>
#include <hip/hip_bf16.h>
#include <math.h>

// ---------------------------------------------------------------------------
// CDNA5 (gfx1250) wave32 WMMA implementation of the VQ-VAE forward pass.
// - convs / projections: implicit-GEMM on v_wmma_f32_16x16x32_bf16, 128-bit
//   operand loads, one A fragment reused by 4 WMMAs per K step.
// - VQ distance+argmin: codebook tiles staged through double-buffered LDS via
//   global_load_async_to_lds_b128 (ASYNCcnt), shared by all 8 waves per block.
// ---------------------------------------------------------------------------

typedef __attribute__((ext_vector_type(16))) __bf16 v16bf;
typedef __attribute__((ext_vector_type(8)))  float  v8f;

union Frag { unsigned u[8]; uint4 q[2]; v16bf bf; };

#define CDIM   128
#define NROWS  8192          // 2 * 64 * 64 spatial rows (NHWC)
#define KCODES 10240
#define HWSZ   4096

__device__ __forceinline__ unsigned short f2bf(float f) {
  unsigned u = __float_as_uint(f);
  u += 0x7FFFu + ((u >> 16) & 1u);          // round-to-nearest-even
  return (unsigned short)(u >> 16);
}

// ---------------------------------------------------------------------------
// Elementwise / layout kernels
// ---------------------------------------------------------------------------

__global__ void k_zero_f(float* p, int n) {
  int t = blockIdx.x * blockDim.x + threadIdx.x;
  if (t < n) p[t] = 0.f;
}
__global__ void k_zero_i(int* p, int n) {
  int t = blockIdx.x * blockDim.x + threadIdx.x;
  if (t < n) p[t] = 0;
}

// x (2,4,64,64) NCHW fp32 -> bf16 NHWC with channel pad to 32
__global__ void k_in_nhwc(const float* __restrict__ x, unsigned short* __restrict__ dst) {
  int t = blockIdx.x * blockDim.x + threadIdx.x;           // 8192*32
  if (t >= NROWS * 32) return;
  int n = t >> 5, c = t & 31;
  int b = n >> 12, s = n & 4095;
  float v = (c < 4) ? x[(size_t)b * 16384 + c * HWSZ + s] : 0.f;
  dst[t] = f2bf(v);
}

__global__ void k_f32_to_bf16(const float* __restrict__ s, unsigned short* __restrict__ d, int n) {
  int t = blockIdx.x * blockDim.x + threadIdx.x;
  if (t < n) d[t] = f2bf(s[t]);
}

// final NHWC(stride 4) -> NCHW (2,4,64,64) into d_out
__global__ void k_out_nchw(const float* __restrict__ nhwc4, float* __restrict__ out) {
  int t = blockIdx.x * blockDim.x + threadIdx.x;           // 32768
  if (t >= 32768) return;
  int b = t >> 14, c = (t >> 12) & 3, s = t & 4095;
  out[t] = nhwc4[((size_t)(b * HWSZ + s)) * 4 + c];
}

// ---------------------------------------------------------------------------
// GroupNorm (32 groups of 4 channels), NHWC fp32
// ---------------------------------------------------------------------------

__global__ void k_gn_stats(const float* __restrict__ act, float* __restrict__ stats) {
  int bid = blockIdx.x;                                    // 0..63  (batch*32 + group)
  int b = bid >> 5, g = bid & 31;
  const float* p = act + (size_t)b * HWSZ * CDIM + g * 4;
  float s = 0.f, ss = 0.f;
  for (int i = threadIdx.x; i < HWSZ; i += blockDim.x) {
    const float4 v4 = *(const float4*)(p + (size_t)i * CDIM);
    s  += v4.x + v4.y + v4.z + v4.w;
    ss += v4.x * v4.x + v4.y * v4.y + v4.z * v4.z + v4.w * v4.w;
  }
  __shared__ float shs[256], shss[256];
  shs[threadIdx.x] = s; shss[threadIdx.x] = ss; __syncthreads();
  for (int o = 128; o > 0; o >>= 1) {
    if (threadIdx.x < o) { shs[threadIdx.x] += shs[threadIdx.x + o]; shss[threadIdx.x] += shss[threadIdx.x + o]; }
    __syncthreads();
  }
  if (threadIdx.x == 0) {
    float m = shs[0] * (1.f / 16384.f);
    float var = shss[0] * (1.f / 16384.f) - m * m;
    stats[bid * 2] = m;
    stats[bid * 2 + 1] = rsqrtf(var + 1e-6f);
  }
}

__global__ void k_gn_apply(const float* __restrict__ act, const float* __restrict__ gamma,
                           const float* __restrict__ beta, const float* __restrict__ stats,
                           int do_swish, unsigned short* __restrict__ bf_out, float* __restrict__ f_out) {
  int t = blockIdx.x * blockDim.x + threadIdx.x;           // 8192*128
  if (t >= NROWS * CDIM) return;
  int c = t & 127, n = t >> 7, b = n >> 12, g = c >> 2;
  float m  = stats[(b * 32 + g) * 2];
  float rs = stats[(b * 32 + g) * 2 + 1];
  float y = (act[t] - m) * rs * gamma[c] + beta[c];
  if (do_swish) y = y / (1.f + __expf(-y));                // x * sigmoid(x)
  if (bf_out) bf_out[t] = f2bf(y);
  if (f_out)  f_out[t]  = y;
}

// ---------------------------------------------------------------------------
// Weight packing into WMMA B-fragment lane layout:
//   dst[((nt*Kb + kb)*32 + lane)*16 + j] holds B[k][n] with
//   n = nt*16 + (lane&15),  k = kb*32 + (lane>=16 ? 8 : 0) + (j&7) + (j>>3)*16
// ---------------------------------------------------------------------------

// conv weight OIHW, K decomposed as k = (ky*3+kx)*Cpad + ci
__global__ void k_pack_conv_w(const float* __restrict__ w, unsigned short* __restrict__ dst,
                              int Cout, int Cin, int Cpad, int Kb, int ntiles) {
  int total = ntiles * Kb * 512;
  int p = blockIdx.x * blockDim.x + threadIdx.x;
  if (p >= total) return;
  int j = p & 15, lane = (p >> 4) & 31, kb = (p >> 9) % Kb, nt = p / (Kb * 512);
  int co = nt * 16 + (lane & 15);
  int K = kb * 32 + ((lane >= 16) ? 8 : 0) + (j & 7) + (j >> 3) * 16;
  int ci = K % Cpad, q = K / Cpad;
  int ky = q / 3, kx = q - ky * 3;
  float v = 0.f;
  if (co < Cout && ci < Cin && q < 9)
    v = w[((size_t)(co * Cin + ci) * 3 + ky) * 3 + kx];
  dst[p] = f2bf(v);
}

// row-major src [N,K] fp32, B[k][n] = src[n][k]  (for sv_w1/sv_w2/codebook)
__global__ void k_pack_b_rm(const float* __restrict__ src, unsigned short* __restrict__ dst, int N, int Kdim) {
  int Kb = Kdim >> 5;
  int total = (N >> 4) * Kb * 512;
  int p = blockIdx.x * blockDim.x + threadIdx.x;
  if (p >= total) return;
  int j = p & 15, lane = (p >> 4) & 31, kb = (p >> 9) % Kb, nt = p / (Kb * 512);
  int n = nt * 16 + (lane & 15);
  int k = kb * 32 + ((lane >= 16) ? 8 : 0) + (j & 7) + (j >> 3) * 16;
  dst[p] = f2bf(src[(size_t)n * Kdim + k]);
}

// ---------------------------------------------------------------------------
// Implicit-GEMM 3x3 SAME conv. One wave computes a 16 x (16*NTW) output strip:
// one A im2col fragment per K step is reused by NTW WMMAs. A/B fragments are
// fetched with 128-bit loads (A-fragment = two aligned 16B chunks at
// K offsets hi*8+{0..7} and hi*8+{16..23}, both within one 3x3 tap).
// ---------------------------------------------------------------------------

template<int CPAD, int KB, int NTW, int NTG>
__global__ void k_conv_wmma(const unsigned short* __restrict__ actbf, const unsigned short* __restrict__ wpack,
                            const float* __restrict__ bias, const float* __restrict__ res,
                            float* __restrict__ out, unsigned short* __restrict__ out_bf,
                            int Cout, int OS) {
  int lane = threadIdx.x & 31;
  int tile = blockIdx.x * (blockDim.x >> 5) + (threadIdx.x >> 5);
  if (tile >= 512 * NTG) return;
  int mt = tile / NTG, ng = tile - mt * NTG;     // NTG is compile-time -> no div
  int hi = lane >> 4;
  int row = mt * 16 + (lane & 15);
  int b = row >> 12, yy = (row >> 6) & 63, xx = row & 63;

  v8f acc[NTW];
#pragma unroll
  for (int t = 0; t < NTW; ++t) acc[t] = (v8f){0.f, 0.f, 0.f, 0.f, 0.f, 0.f, 0.f, 0.f};

  for (int kb = 0; kb < KB; ++kb) {
    Frag a;
#pragma unroll
    for (int h = 0; h < 2; ++h) {
      int K0 = kb * 32 + hi * 8 + h * 16;
      int ci = K0 % CPAD, q = K0 / CPAD;         // compile-time CPAD -> shifts
      int ky = q / 3, kx = q - ky * 3;
      int iy = yy + ky - 1, ix = xx + kx - 1;
      uint4 val = {0u, 0u, 0u, 0u};
      if ((unsigned)iy < 64u && (unsigned)ix < 64u)
        val = *(const uint4*)(actbf + ((size_t)(b * HWSZ + iy * 64 + ix)) * CPAD + ci);
      a.q[h] = val;
    }
#pragma unroll
    for (int t = 0; t < NTW; ++t) {
      int nt = ng * NTW + t;
      const uint4* wp = (const uint4*)((const unsigned*)wpack + ((size_t)(nt * KB + kb) * 32 + lane) * 8);
      Frag bw;
      bw.q[0] = wp[0];
      bw.q[1] = wp[1];
      acc[t] = __builtin_amdgcn_wmma_f32_16x16x32_bf16(false, a.bf, false, bw.bf,
                                                       (short)0, acc[t], false, false);
    }
  }
#pragma unroll
  for (int t = 0; t < NTW; ++t) {
    int col = (ng * NTW + t) * 16 + (lane & 15);
    if (col < Cout) {
      float bv = bias ? bias[col] : 0.f;
#pragma unroll
      for (int v = 0; v < 8; ++v) {
        int r = mt * 16 + v + hi * 8;
        float o = acc[t][v] + bv;
        if (res) o += res[(size_t)r * CDIM + col];
        out[(size_t)r * OS + col] = o;
        if (out_bf) out_bf[(size_t)r * OS + col] = f2bf(o);
      }
    }
  }
}

// ---------------------------------------------------------------------------
// Generic GEMM: A bf16 row-major [M, KB*32], B prepacked, C fp32 [M, NTG*NTW*16]
// optional exact GELU on output, optional bf16 mirror of the output.
// ---------------------------------------------------------------------------

template<int KB, int NTW, int NTG>
__global__ void k_gemm_wmma(const unsigned short* __restrict__ Abf, const unsigned short* __restrict__ Bpack,
                            float* __restrict__ out, unsigned short* __restrict__ out_bf,
                            int M, int act_gelu) {
  int lane = threadIdx.x & 31;
  int tile = blockIdx.x * (blockDim.x >> 5) + (threadIdx.x >> 5);
  int mtiles = M >> 4;
  if (tile >= mtiles * NTG) return;
  int mt = tile / NTG, ng = tile - mt * NTG;
  int hi = lane >> 4;
  int row = mt * 16 + (lane & 15);
  const int Kdim = KB * 32;

  v8f acc[NTW];
#pragma unroll
  for (int t = 0; t < NTW; ++t) acc[t] = (v8f){0.f, 0.f, 0.f, 0.f, 0.f, 0.f, 0.f, 0.f};

  for (int kb = 0; kb < KB; ++kb) {
    Frag a;
    const unsigned short* ap = Abf + (size_t)row * Kdim + kb * 32 + hi * 8;
    a.q[0] = *(const uint4*)(ap);
    a.q[1] = *(const uint4*)(ap + 16);
#pragma unroll
    for (int t = 0; t < NTW; ++t) {
      int nt = ng * NTW + t;
      const uint4* wp = (const uint4*)((const unsigned*)Bpack + ((size_t)(nt * KB + kb) * 32 + lane) * 8);
      Frag bw;
      bw.q[0] = wp[0];
      bw.q[1] = wp[1];
      acc[t] = __builtin_amdgcn_wmma_f32_16x16x32_bf16(false, a.bf, false, bw.bf,
                                                       (short)0, acc[t], false, false);
    }
  }
  const int Ncols = NTG * NTW * 16;
#pragma unroll
  for (int t = 0; t < NTW; ++t) {
    int col = (ng * NTW + t) * 16 + (lane & 15);
#pragma unroll
    for (int v = 0; v < 8; ++v) {
      int r = mt * 16 + v + hi * 8;
      float o = acc[t][v];
      if (act_gelu) o = 0.5f * o * (1.f + erff(o * 0.70710678118f));
      out[(size_t)r * Ncols + col] = o;
      if (out_bf) out_bf[(size_t)r * Ncols + col] = f2bf(o);
    }
  }
}

// ---------------------------------------------------------------------------
// Fused distance + argmin. All 8 waves of a block sweep the SAME 640 codebook
// tiles, so each contiguous 4KB tile is staged once per block into a
// double-buffered LDS buffer with global_load_async_to_lds_b128 (ASYNCcnt),
// then every wave reads its B fragments from LDS (ds_load_b128).
// d = |z|^2 + |cb|^2 - 2*(z . cb); distance matrix never materialized.
// Grid must be exactly 64 blocks x 256 threads (all threads hit the barriers).
// ---------------------------------------------------------------------------

__global__ void k_vq_argmin(const unsigned short* __restrict__ zbf, const unsigned short* __restrict__ cbpack,
                            const float* __restrict__ znorm, const float* __restrict__ cbnorm,
                            int* __restrict__ idx) {
  __shared__ unsigned short shB[2][2048];                  // 2 x 4KB code tiles
  const int NT = KCODES / 16;                              // 640 tiles
  int lane = threadIdx.x & 31;
  int mt = blockIdx.x * (blockDim.x >> 5) + (threadIdx.x >> 5);   // 0..511 (exact)
  int hi = lane >> 4;
  int row = mt * 16 + (lane & 15);

  // async stage of one 4KB tile: 256 threads x 16B
  auto fetch = [&](int ct, int buf) {
    const unsigned short* g = cbpack + (size_t)ct * 2048 + threadIdx.x * 8;
    unsigned lds = (unsigned)(uintptr_t)(&shB[buf][threadIdx.x * 8]);
    asm volatile("global_load_async_to_lds_b128 %0, %1, off"
                 :: "v"(lds), "v"(g) : "memory");
  };

  Frag a[4];
#pragma unroll
  for (int kb = 0; kb < 4; ++kb) {
    const unsigned short* ap = zbf + (size_t)row * CDIM + kb * 32 + hi * 8;
    a[kb].q[0] = *(const uint4*)(ap);
    a[kb].q[1] = *(const uint4*)(ap + 16);
  }

  float bestd[8]; int besti[8]; float myzn[8];
#pragma unroll
  for (int v = 0; v < 8; ++v) {
    bestd[v] = 3.4e38f; besti[v] = 0;
    myzn[v] = znorm[mt * 16 + v + hi * 8];
  }

  fetch(0, 0);
  for (int ct = 0; ct < NT; ++ct) {
    int cur = ct & 1;
    if (ct + 1 < NT) {
      fetch(ct + 1, cur ^ 1);
      asm volatile("s_wait_asynccnt 0x1" ::: "memory");    // tile ct landed, ct+1 in flight
    } else {
      asm volatile("s_wait_asynccnt 0x0" ::: "memory");
    }
    __syncthreads();                                       // all waves' copy portions visible

    v8f acc = {0.f, 0.f, 0.f, 0.f, 0.f, 0.f, 0.f, 0.f};
#pragma unroll
    for (int kb = 0; kb < 4; ++kb) {
      Frag bw;
      const uint4* lp = (const uint4*)(&shB[cur][(kb * 32 + lane) * 16]);
      bw.q[0] = lp[0];
      bw.q[1] = lp[1];
      acc = __builtin_amdgcn_wmma_f32_16x16x32_bf16(false, a[kb].bf, false, bw.bf,
                                                    (short)0, acc, false, false);
    }
    int code = ct * 16 + (lane & 15);
    float cn = cbnorm[code];
#pragma unroll
    for (int v = 0; v < 8; ++v) {
      float d = myzn[v] + cn - 2.f * acc[v];
      if (d < bestd[v]) { bestd[v] = d; besti[v] = code; }
    }
    __syncthreads();                                       // protect buf before next overwrite
  }
  // reduce (min, argmin) across the 16 lanes of each half (same row per half)
#pragma unroll
  for (int v = 0; v < 8; ++v) {
    float bd = bestd[v]; int bi = besti[v];
#pragma unroll
    for (int o = 8; o >= 1; o >>= 1) {
      float od = __shfl_xor(bd, o, 32);
      int   oi = __shfl_xor(bi, o, 32);
      if (od < bd || (od == bd && oi < bi)) { bd = od; bi = oi; }
    }
    if ((lane & 15) == 0) idx[mt * 16 + v + hi * 8] = bi;
  }
}

// ---------------------------------------------------------------------------
// Reductions / VQ bookkeeping
// ---------------------------------------------------------------------------

__global__ void k_rownorm(const float* __restrict__ src, float* __restrict__ nrm) { // K=128, grid=M, block=128
  int r = blockIdx.x;
  __shared__ float sh[128];
  float v = src[(size_t)r * CDIM + threadIdx.x];
  sh[threadIdx.x] = v * v; __syncthreads();
  for (int o = 64; o > 0; o >>= 1) {
    if (threadIdx.x < o) sh[threadIdx.x] += sh[threadIdx.x + o];
    __syncthreads();
  }
  if (threadIdx.x == 0) nrm[r] = sh[0];
}

__global__ void k_colsum(const float* __restrict__ src, float* __restrict__ col, int M) { // grid=128
  int c = blockIdx.x;
  float s = 0.f;
  for (int r = threadIdx.x; r < M; r += blockDim.x) s += src[(size_t)r * CDIM + c];
  __shared__ float sh[256];
  sh[threadIdx.x] = s; __syncthreads();
  for (int o = 128; o > 0; o >>= 1) {
    if (threadIdx.x < o) sh[threadIdx.x] += sh[threadIdx.x + o];
    __syncthreads();
  }
  if (threadIdx.x == 0) col[c] = sh[0];
}

__global__ void k_sum(const float* __restrict__ src, int n, float* __restrict__ acc) {
  float s = 0.f;
  for (int i = blockIdx.x * blockDim.x + threadIdx.x; i < n; i += gridDim.x * blockDim.x) s += src[i];
  __shared__ float sh[256];
  sh[threadIdx.x] = s; __syncthreads();
  for (int o = 128; o > 0; o >>= 1) {
    if (threadIdx.x < o) sh[threadIdx.x] += sh[threadIdx.x + o];
    __syncthreads();
  }
  if (threadIdx.x == 0) atomicAdd(acc, sh[0]);
}

__global__ void k_gather_loss(const float* __restrict__ cb, const int* __restrict__ idx,
                              const float* __restrict__ z, float* __restrict__ dec,
                              float* __restrict__ loss_acc) {
  int t = blockIdx.x * blockDim.x + threadIdx.x;           // 8192*128 exactly
  int n = t >> 7, c = t & 127;
  float q = cb[(size_t)idx[n] * CDIM + c];
  dec[t] = q;
  float df = q - z[t];
  __shared__ float sh[256];
  sh[threadIdx.x] = df * df; __syncthreads();
  for (int o = 128; o > 0; o >>= 1) {
    if (threadIdx.x < o) sh[threadIdx.x] += sh[threadIdx.x + o];
    __syncthreads();
  }
  if (threadIdx.x == 0) atomicAdd(loss_acc, sh[0]);
}

__global__ void k_hist(const int* __restrict__ idx, int* __restrict__ hist) {
  int t = blockIdx.x * blockDim.x + threadIdx.x;
  if (t < NROWS) atomicAdd(&hist[idx[t]], 1);
}

__global__ void k_finalize(const int* __restrict__ hist, const float* __restrict__ loss_acc,
                           const float* __restrict__ szn, const float* __restrict__ scbn,
                           const float* __restrict__ colz, const float* __restrict__ colcb,
                           float* __restrict__ outs) {
  __shared__ float sh[256];
  int tid = threadIdx.x;
  float s = 0.f;
  for (int k = tid; k < KCODES; k += 256) {
    float e = (float)hist[k] * (1.f / (float)NROWS);
    s += e * __logf(e + 1e-10f);
  }
  sh[tid] = s; __syncthreads();
  for (int o = 128; o > 0; o >>= 1) { if (tid < o) sh[tid] += sh[tid + o]; __syncthreads(); }
  float ent = sh[0]; __syncthreads();
  sh[tid] = (tid < CDIM) ? colz[tid] * colcb[tid] : 0.f; __syncthreads();
  for (int o = 128; o > 0; o >>= 1) { if (tid < o) sh[tid] += sh[tid + o]; __syncthreads(); }
  if (tid == 0) {
    outs[0] = loss_acc[0] * (1.25f / ((float)NROWS * (float)CDIM));             // loss
    outs[1] = __expf(-ent);                                                     // perplexity
    outs[2] = ((float)KCODES * szn[0] + (float)NROWS * scbn[0] - 2.f * sh[0])
              * (1.f / ((float)NROWS * (float)KCODES));                         // mean_distance
  }
}

// ---------------------------------------------------------------------------
// Host orchestration
// ---------------------------------------------------------------------------

extern "C" void kernel_launch(void* const* d_in, const int* in_sizes, int n_in,
                              void* d_out, int out_size, void* d_ws, size_t ws_size,
                              hipStream_t stream) {
  (void)in_sizes; (void)n_in; (void)out_size; (void)ws_size;

  const float* x     = (const float*)d_in[0];
  const float* w_in  = (const float*)d_in[1];
  const float* c_in  = (const float*)d_in[2];
  const float* g_in  = (const float*)d_in[3];
  const float* b_in  = (const float*)d_in[4];
  const float* g_enc = (const float*)d_in[53];
  const float* b_enc = (const float*)d_in[54];
  const float* w_enc = (const float*)d_in[55];
  const float* c_enc = (const float*)d_in[56];
  const float* emb   = (const float*)d_in[57];
  const float* sv_w1 = (const float*)d_in[58];
  const float* sv_w2 = (const float*)d_in[59];
  const float* g_out = (const float*)d_in[108];
  const float* b_out = (const float*)d_in[109];
  const float* w_out = (const float*)d_in[110];
  const float* c_out = (const float*)d_in[111];

  // ---- workspace carve-up -------------------------------------------------
  size_t off = 0;
  auto carve = [&](size_t bytes) -> void* {
    void* p = (char*)d_ws + off;
    off += (bytes + 255) & ~(size_t)255;
    return p;
  };
  unsigned short* inbf  = (unsigned short*)carve((size_t)NROWS * 32 * 2);
  float* actA = (float*)carve((size_t)NROWS * CDIM * 4);
  float* actB = (float*)carve((size_t)NROWS * CDIM * 4);
  float* actZ = (float*)carve((size_t)NROWS * CDIM * 4);
  unsigned short* bfbuf = (unsigned short*)carve((size_t)NROWS * CDIM * 2);
  unsigned short* zbf   = (unsigned short*)carve((size_t)NROWS * CDIM * 2);
  unsigned short* wp_in  = (unsigned short*)carve((size_t)8 * 9 * 512 * 2);
  unsigned short* wp_enc = (unsigned short*)carve((size_t)8 * 36 * 512 * 2);
  unsigned short* wp_out = (unsigned short*)carve((size_t)1 * 36 * 512 * 2);
  unsigned short* wp_blk[24];
  for (int i = 0; i < 24; ++i) wp_blk[i] = (unsigned short*)carve((size_t)8 * 36 * 512 * 2);
  unsigned short* svp1  = (unsigned short*)carve((size_t)8 * 4 * 512 * 2);
  unsigned short* svp2  = (unsigned short*)carve((size_t)8 * 4 * 512 * 2);
  unsigned short* embbf = (unsigned short*)carve((size_t)KCODES * CDIM * 2);
  float* tbuf = (float*)carve((size_t)KCODES * CDIM * 4);
  unsigned short* tbf   = (unsigned short*)carve((size_t)KCODES * CDIM * 2);
  float* cbf  = (float*)carve((size_t)KCODES * CDIM * 4);
  unsigned short* cbpack = (unsigned short*)carve((size_t)KCODES * CDIM * 2);
  float* znorm  = (float*)carve((size_t)NROWS * 4);
  float* cbnorm = (float*)carve((size_t)KCODES * 4);
  float* colz   = (float*)carve(CDIM * 4);
  float* colcb  = (float*)carve(CDIM * 4);
  int*   idx    = (int*)carve((size_t)NROWS * 4);
  int*   hist   = (int*)carve((size_t)KCODES * 4);
  float* stats  = (float*)carve(64 * 2 * 4);
  float* scal   = (float*)carve(16 * 4);      // [0]=loss_acc [1]=sum znorm [2]=sum cbnorm
  float* outh4  = (float*)carve((size_t)NROWS * 4 * 4);

  float* out = (float*)d_out;
  unsigned short* nobf = nullptr;

  // ---- init accumulators --------------------------------------------------
  k_zero_f<<<1, 16, 0, stream>>>(scal, 16);
  k_zero_i<<<(KCODES + 255) / 256, 256, 0, stream>>>(hist, KCODES);

  // ---- pack all conv / linear weights into WMMA B layout ------------------
  k_pack_conv_w<<<(8 * 9 * 512 + 255) / 256, 256, 0, stream>>>(w_in, wp_in, 128, 4, 32, 9, 8);
  k_pack_conv_w<<<(8 * 36 * 512 + 255) / 256, 256, 0, stream>>>(w_enc, wp_enc, 128, 128, 128, 36, 8);
  k_pack_conv_w<<<(1 * 36 * 512 + 255) / 256, 256, 0, stream>>>(w_out, wp_out, 4, 128, 128, 36, 1);
  for (int i = 0; i < 6; ++i) {
    const float* w1e = (const float*)d_in[5 + i * 8 + 2];
    const float* w2e = (const float*)d_in[5 + i * 8 + 6];
    const float* w1d = (const float*)d_in[60 + i * 8 + 2];
    const float* w2d = (const float*)d_in[60 + i * 8 + 6];
    k_pack_conv_w<<<(8 * 36 * 512 + 255) / 256, 256, 0, stream>>>(w1e, wp_blk[i * 2 + 0], 128, 128, 128, 36, 8);
    k_pack_conv_w<<<(8 * 36 * 512 + 255) / 256, 256, 0, stream>>>(w2e, wp_blk[i * 2 + 1], 128, 128, 128, 36, 8);
    k_pack_conv_w<<<(8 * 36 * 512 + 255) / 256, 256, 0, stream>>>(w1d, wp_blk[12 + i * 2 + 0], 128, 128, 128, 36, 8);
    k_pack_conv_w<<<(8 * 36 * 512 + 255) / 256, 256, 0, stream>>>(w2d, wp_blk[12 + i * 2 + 1], 128, 128, 128, 36, 8);
  }
  k_pack_b_rm<<<(8 * 4 * 512 + 255) / 256, 256, 0, stream>>>(sv_w1, svp1, 128, 128);
  k_pack_b_rm<<<(8 * 4 * 512 + 255) / 256, 256, 0, stream>>>(sv_w2, svp2, 128, 128);

  // ---- resblock helper ----------------------------------------------------
  auto resblock = [&](float* actX, int base) {
    const float* g1 = (const float*)d_in[base + 0];
    const float* b1 = (const float*)d_in[base + 1];
    const float* c1 = (const float*)d_in[base + 3];
    const float* g2 = (const float*)d_in[base + 4];
    const float* b2 = (const float*)d_in[base + 5];
    const float* c2 = (const float*)d_in[base + 7];
    int wi = (base >= 60) ? (12 + ((base - 60) / 8) * 2) : (((base - 5) / 8) * 2);
    k_gn_stats<<<64, 256, 0, stream>>>(actX, stats);
    k_gn_apply<<<4096, 256, 0, stream>>>(actX, g1, b1, stats, 1, bfbuf, nullptr);
    k_conv_wmma<128, 36, 4, 2><<<128, 256, 0, stream>>>(bfbuf, wp_blk[wi + 0], c1, nullptr, actB, nobf, 128, 128);
    k_gn_stats<<<64, 256, 0, stream>>>(actB, stats);
    k_gn_apply<<<4096, 256, 0, stream>>>(actB, g2, b2, stats, 1, bfbuf, nullptr);
    k_conv_wmma<128, 36, 4, 2><<<128, 256, 0, stream>>>(bfbuf, wp_blk[wi + 1], c2, actX, actX, nobf, 128, 128);
  };

  // ---- encoder ------------------------------------------------------------
  k_in_nhwc<<<(NROWS * 32 + 255) / 256, 256, 0, stream>>>(x, inbf);
  k_conv_wmma<32, 9, 4, 2><<<128, 256, 0, stream>>>(inbf, wp_in, c_in, nullptr, actA, nobf, 128, 128);
  k_gn_stats<<<64, 256, 0, stream>>>(actA, stats);
  k_gn_apply<<<4096, 256, 0, stream>>>(actA, g_in, b_in, stats, 0, nullptr, actA);
  for (int i = 0; i < 6; ++i) resblock(actA, 5 + i * 8);
  k_gn_stats<<<64, 256, 0, stream>>>(actA, stats);
  k_gn_apply<<<4096, 256, 0, stream>>>(actA, g_enc, b_enc, stats, 0, bfbuf, nullptr);
  k_conv_wmma<128, 36, 4, 2><<<128, 256, 0, stream>>>(bfbuf, wp_enc, c_enc, nullptr, actZ, zbf, 128, 128);

  // ---- vector quantizer ---------------------------------------------------
  k_rownorm<<<NROWS, 128, 0, stream>>>(actZ, znorm);
  k_colsum<<<CDIM, 256, 0, stream>>>(actZ, colz, NROWS);
  k_sum<<<32, 256, 0, stream>>>(znorm, NROWS, scal + 1);

  k_f32_to_bf16<<<(KCODES * CDIM + 255) / 256, 256, 0, stream>>>(emb, embbf, KCODES * CDIM);
  k_gemm_wmma<4, 4, 2><<<160, 256, 0, stream>>>(embbf, svp1, tbuf, tbf, KCODES, 1);   // gelu(emb @ w1.T)
  k_gemm_wmma<4, 4, 2><<<160, 256, 0, stream>>>(tbf, svp2, cbf, nobf, KCODES, 0);     // @ w2.T
  k_rownorm<<<KCODES, 128, 0, stream>>>(cbf, cbnorm);
  k_colsum<<<CDIM, 256, 0, stream>>>(cbf, colcb, KCODES);
  k_sum<<<32, 256, 0, stream>>>(cbnorm, KCODES, scal + 2);
  k_pack_b_rm<<<(640 * 4 * 512 + 255) / 256, 256, 0, stream>>>(cbf, cbpack, KCODES, 128);

  k_vq_argmin<<<64, 256, 0, stream>>>(zbf, cbpack, znorm, cbnorm, idx);
  k_hist<<<(NROWS + 255) / 256, 256, 0, stream>>>(idx, hist);
  k_gather_loss<<<4096, 256, 0, stream>>>(cbf, idx, actZ, actA, scal + 0);   // quant -> actA

  // ---- decoder ------------------------------------------------------------
  for (int i = 0; i < 6; ++i) resblock(actA, 60 + i * 8);
  k_gn_stats<<<64, 256, 0, stream>>>(actA, stats);
  k_gn_apply<<<4096, 256, 0, stream>>>(actA, g_out, b_out, stats, 0, bfbuf, nullptr);
  k_conv_wmma<128, 36, 1, 1><<<64, 256, 0, stream>>>(bfbuf, wp_out, c_out, nullptr, outh4, nobf, 4, 4);

  // ---- outputs ------------------------------------------------------------
  k_out_nchw<<<128, 256, 0, stream>>>(outh4, out);
  k_finalize<<<1, 256, 0, stream>>>(hist, scal + 0, scal + 1, scal + 2, colz, colcb, out + 32768);
}
